// GAT_Multi_77919296684445
// MI455X (gfx1250) — matrix operand order repaired and checked by reading, not compiled, not run on hardware
//
#include <hip/hip_runtime.h>

#define B_ 16
#define N_ 1024
#define F_ 128
#define H_ 256
#define NEGV (-9.0e15f)

typedef __attribute__((ext_vector_type(16))) __bf16 v16bf;
typedef __attribute__((ext_vector_type(8)))  __bf16 v8bf;
typedef __attribute__((ext_vector_type(8)))  float  v8f;
typedef __attribute__((__vector_size__(16))) int    v4i_;
typedef __attribute__((address_space(1))) v4i_* gv4i_p;  // global int4*
typedef __attribute__((address_space(3))) v4i_* lv4i_p;  // LDS int4*

__device__ __forceinline__ float red16(float v) {
#pragma unroll
  for (int off = 1; off < 16; off <<= 1) v += __shfl_xor(v, off, 32);
  return v;
}
__device__ __forceinline__ float red32(float v) {
#pragma unroll
  for (int off = 16; off >= 1; off >>= 1) v += __shfl_xor(v, off, 32);
  return v;
}

// ---------- tiny prep: collapse W@a projections (layer1 src/dst, layer2 all) ----------
__global__ void k_prep_vec(const float* __restrict__ W1, const float* __restrict__ a1,
                           const float* __restrict__ W2, const float* __restrict__ a2,
                           float* __restrict__ w1s, float* __restrict__ w1d,
                           float* __restrict__ w2s, float* __restrict__ w2d,
                           float* __restrict__ w2c) {
  int t = threadIdx.x;
  if (t < F_) {
    float s = 0.f, d = 0.f;
    for (int h = 0; h < H_; ++h) {
      float w = W1[t * H_ + h];
      s += w * a1[h];
      d += w * a1[H_ + h];
    }
    w1s[t] = s; w1d[t] = d;
  }
  if (t < H_) {
    float s = 0.f, d = 0.f;
    for (int f = 0; f < F_; ++f) {
      float w = W2[t * F_ + f];
      s += w * a2[f];
      d += w * a2[F_ + f];
    }
    w2s[t] = s; w2d[t] = d; w2c[t] = W2[t * F_ + (F_ - 1)];
  }
}

// ---------- casts ----------
__global__ void k_cast_x(const float* __restrict__ x, __bf16* __restrict__ xb, int n) {
  int i = blockIdx.x * blockDim.x + threadIdx.x;
  if (i < n) xb[i] = (__bf16)x[i];
}

__global__ void k_w1t(const float* __restrict__ W1, __bf16* __restrict__ w1t) {
  int i = blockIdx.x * blockDim.x + threadIdx.x;  // over H_*F_
  int h = i / F_, f = i % F_;
  w1t[i] = (__bf16)W1[f * H_ + h];                // (H,F) = W1^T in bf16
}

// ---------- adjacency -> bitmask (4 MB -> 128 KB of HBM traffic per pass) ----------
__global__ void k_maskpack(const int* __restrict__ adj, unsigned* __restrict__ mb) {
  int idx = blockIdx.x * blockDim.x + threadIdx.x;  // N_*(N_/32)
  int i = idx >> 5, wcol = idx & 31;
  const int* p = adj + (size_t)i * N_ + wcol * 32;
  unsigned word = 0;
#pragma unroll 8
  for (int b = 0; b < 32; ++b) word |= (p[b] > 0 ? 1u : 0u) << b;
  mb[idx] = word;
}

// ---------- src1/dst1 = x @ (W1 a1) : one wave per row ----------
__global__ void k_srcdst1(const float* __restrict__ x, const float* __restrict__ w1s,
                          const float* __restrict__ w1d, float* __restrict__ src1,
                          float* __restrict__ dst1) {
  int wid = threadIdx.x >> 5, lane = threadIdx.x & 31;
  int r = blockIdx.x * 8 + wid;
  float4 xv = *(const float4*)(x + (size_t)r * F_ + lane * 4);
  float4 ws = *(const float4*)(w1s + lane * 4);
  float4 wd = *(const float4*)(w1d + lane * 4);
  float s = xv.x * ws.x + xv.y * ws.y + xv.z * ws.z + xv.w * ws.w;
  float d = xv.x * wd.x + xv.y * wd.y + xv.z * wd.z + xv.w * wd.w;
  s = red32(s); d = red32(d);
  if (lane == 0) { src1[r] = s; dst1[r] = d; }
}

// ---------- pass A: per-row softmax stats for layer-1 (rank-1 logits, never stored) ----------
__global__ void k_stats1(const float* __restrict__ src1, const float* __restrict__ dst1,
                         const unsigned* __restrict__ mb,
                         float* __restrict__ m1, float* __restrict__ s1inv) {
  int wid = threadIdx.x >> 5, lane = threadIdx.x & 31;
  int g = blockIdx.x * 8 + wid;
  int b = g >> 10, i = g & (N_ - 1);
  float sv = src1[g];
  const float* dB = dst1 + b * N_;
  const unsigned* mrow = mb + (size_t)i * (N_ / 32);
  float m = -3.4e38f, s = 0.f;
  for (int it = 0; it < N_ / 32; ++it) {
    float e = sv + dB[it * 32 + lane];
    e = e > 0.f ? e : 0.2f * e;
    e = ((mrow[it] >> lane) & 1u) ? e : NEGV;
    float mn = fmaxf(m, e);
    float sc = __expf(m - mn);
    s = s * sc + __expf(e - mn);
    m = mn;
  }
#pragma unroll
  for (int off = 16; off >= 1; off >>= 1) {
    float mo = __shfl_xor(m, off, 32);
    float so = __shfl_xor(s, off, 32);
    float mn = fmaxf(m, mo);
    s = s * __expf(m - mn) + so * __expf(mo - mn);
    m = mn;
  }
  if (lane == 0) { m1[g] = m; s1inv[g] = 1.f / s; }
}

// ---------- GEMM h^T = (x@W1)^T via bf16 WMMA; output transposed (B,H,N) bf16 ----------
__global__ __launch_bounds__(128) void k_gemm1(const __bf16* __restrict__ xb,
                                               const __bf16* __restrict__ w1t,
                                               __bf16* __restrict__ h1bT) {
  int wid = threadIdx.x >> 5, lane = threadIdx.x & 31;
  int wave = blockIdx.x * 4 + wid;
  int ct = wave & 15, nt = (wave >> 4) & 63, b = wave >> 10;
  int lane15 = lane & 15, hi = lane >> 4;
  int n0 = nt * 16, c0 = ct * 16;
  const __bf16* Arow = xb + (size_t)(b * N_ + n0 + lane15) * F_;
  const __bf16* Brow = w1t + (size_t)(c0 + lane15) * F_;
  v8f c = {0.f, 0.f, 0.f, 0.f, 0.f, 0.f, 0.f, 0.f};
#pragma unroll
  for (int kk = 0; kk < 4; ++kk) {
    int k = kk * 32;
    v8bf alo = *(const v8bf*)(Arow + k + hi * 8);
    v8bf ahi = *(const v8bf*)(Arow + k + 16 + hi * 8);
    v16bf a = __builtin_shufflevector(alo, ahi, 0, 1, 2, 3, 4, 5, 6, 7,
                                      8, 9, 10, 11, 12, 13, 14, 15);
    v8bf blo = *(const v8bf*)(Brow + k + hi * 8);
    v8bf bhi = *(const v8bf*)(Brow + k + 16 + hi * 8);
    v16bf bf = __builtin_shufflevector(blo, bhi, 0, 1, 2, 3, 4, 5, 6, 7,
                                       8, 9, 10, 11, 12, 13, 14, 15);
    c = __builtin_amdgcn_wmma_f32_16x16x32_bf16(false, a, false, bf, (short)0, c,
                                                false, false);
  }
  v8bf o;
#pragma unroll
  for (int v = 0; v < 8; ++v) o[v] = (__bf16)c[v];
  // lane's column c0+lane15 is a row of h^T; rows n0+8*hi..+7 are contiguous
  *(v8bf*)(h1bT + (size_t)(b * H_ + c0 + lane15) * N_ + n0 + 8 * hi) = o;
}

// ---------- fused: flash attn@h (WMMA) -> elu -> LayerNorm -> collapsed layer-2 GEMVs ----
// 4 waves/block; each wave owns 4 channel tiles so one A-fragment (exp work) feeds
// 4 back-to-back WMMAs. dst1 row block staged in LDS via async-to-LDS when available.
__global__ __launch_bounds__(128) void k_attn1(
    const __bf16* __restrict__ h1bT, const float* __restrict__ src1,
    const float* __restrict__ dst1, const float* __restrict__ m1,
    const float* __restrict__ s1inv, const unsigned* __restrict__ mb,
    const float* __restrict__ gamma, const float* __restrict__ beta,
    const float* __restrict__ w2s, const float* __restrict__ w2d,
    const float* __restrict__ w2c, float* __restrict__ src2,
    float* __restrict__ dst2, float* __restrict__ colv) {
  __shared__ __align__(16) float sDst[N_];
  __shared__ float sSum[4][16], sSq[4][16], sPs[4][16], sPd[4][16], sPc[4][16];
  __shared__ float sMu[16], sRstd[16];
  int wid = threadIdx.x >> 5, lane = threadIdx.x & 31;
  int lane15 = lane & 15, hi = lane >> 4;
  int b = blockIdx.x >> 6, i0 = (blockIdx.x & 63) * 16;
  int row = i0 + lane15;
  int g = b * N_ + row;
  float sv = src1[g], mv = m1[g], siv = s1inv[g];

  // stage the 4KB dst1 block for this batch into LDS (async path if toolchain has it)
  const float* gd = dst1 + b * N_;
#if __has_builtin(__builtin_amdgcn_global_load_async_to_lds_b128)
  for (int t = threadIdx.x; t < N_ / 4; t += 128) {
    __builtin_amdgcn_global_load_async_to_lds_b128(
        (gv4i_p)(gd + 4 * t), (lv4i_p)&sDst[4 * t], 0, 0);
  }
#if __has_builtin(__builtin_amdgcn_s_wait_asynccnt)
  __builtin_amdgcn_s_wait_asynccnt(0);
#endif
#else
  for (int t = threadIdx.x; t < N_ / 4; t += 128)
    *(float4*)&sDst[4 * t] = *(const float4*)(gd + 4 * t);
#endif
  __syncthreads();

  // 4 channel tiles per wave: c_u = wid*64 + u*16 + lane15
  const __bf16* Bb[4];
  float cg[4], cb[4], vws[4], vwd[4], vwc[4];
#pragma unroll
  for (int u = 0; u < 4; ++u) {
    int c = wid * 64 + u * 16 + lane15;
    Bb[u] = h1bT + (size_t)(b * H_ + c) * N_;
    cg[u] = gamma[c]; cb[u] = beta[c];
    vws[u] = w2s[c];  vwd[u] = w2d[c]; vwc[u] = w2c[c];
  }
  const unsigned* mrow = mb + (size_t)row * (N_ / 32);
  v8f acc[4];
#pragma unroll
  for (int u = 0; u < 4; ++u) acc[u] = (v8f){0.f, 0.f, 0.f, 0.f, 0.f, 0.f, 0.f, 0.f};

  for (int j0 = 0; j0 < N_; j0 += 32) {
    int jA = j0 + hi * 8;
    float4 q0 = *(const float4*)&sDst[jA];
    float4 q1 = *(const float4*)&sDst[jA + 4];
    float4 q2 = *(const float4*)&sDst[jA + 16];
    float4 q3 = *(const float4*)&sDst[jA + 20];
    float dv[16] = {q0.x, q0.y, q0.z, q0.w, q1.x, q1.y, q1.z, q1.w,
                    q2.x, q2.y, q2.z, q2.w, q3.x, q3.y, q3.z, q3.w};
    unsigned mw = mrow[j0 >> 5];
    v16bf af;
#pragma unroll
    for (int t = 0; t < 16; ++t) {
      int koff = hi * 8 + ((t < 8) ? t : (8 + t));  // column index within 32-wide step
      float e = sv + dv[t];
      e = e > 0.f ? e : 0.2f * e;
      e = ((mw >> koff) & 1u) ? e : NEGV;
      af[t] = (__bf16)(__expf(e - mv) * siv);  // normalized attn prob, bf16
    }
#pragma unroll
    for (int u = 0; u < 4; ++u) {
      v8bf blo = *(const v8bf*)(Bb[u] + j0 + hi * 8);
      v8bf bhi = *(const v8bf*)(Bb[u] + j0 + 16 + hi * 8);
      v16bf bf = __builtin_shufflevector(blo, bhi, 0, 1, 2, 3, 4, 5, 6, 7,
                                         8, 9, 10, 11, 12, 13, 14, 15);
      acc[u] = __builtin_amdgcn_wmma_f32_16x16x32_bf16(false, af, false, bf, (short)0,
                                                       acc[u], false, false);
    }
    __builtin_prefetch((const void*)(Bb[0] + j0 + 64), 0, 1);
  }

  // elu + per-row sums for LayerNorm (sum 4 tiles locally, 16 lanes by shuffle, 4 waves via LDS)
#pragma unroll
  for (int v = 0; v < 8; ++v) {
    float rs = 0.f, rq = 0.f;
#pragma unroll
    for (int u = 0; u < 4; ++u) {
      float xx = acc[u][v];
      xx = xx > 0.f ? xx : (__expf(xx) - 1.f);
      acc[u][v] = xx;
      rs += xx;
      rq += xx * xx;
    }
    rs = red16(rs);
    rq = red16(rq);
    if (lane15 == 0) { sSum[wid][v + 8 * hi] = rs; sSq[wid][v + 8 * hi] = rq; }
  }
  __syncthreads();
  if (threadIdx.x < 16) {
    int r = threadIdx.x;
    float ms = 0.f, qs = 0.f;
    for (int w = 0; w < 4; ++w) { ms += sSum[w][r]; qs += sSq[w][r]; }
    float mu = ms * (1.f / H_);
    float var = qs * (1.f / H_) - mu * mu;
    sMu[r] = mu;
    sRstd[r] = rsqrtf(var + 1e-5f);
  }
  __syncthreads();
#pragma unroll
  for (int v = 0; v < 8; ++v) {
    int r = v + 8 * hi;
    float mu = sMu[r], rstd = sRstd[r];
    float ts = 0.f, td = 0.f, tc = 0.f;
#pragma unroll
    for (int u = 0; u < 4; ++u) {
      float nv = (acc[u][v] - mu) * rstd * cg[u] + cb[u];
      ts += nv * vws[u];
      td += nv * vwd[u];
      tc += nv * vwc[u];
    }
    ts = red16(ts);
    td = red16(td);
    tc = red16(tc);
    if (lane15 == 0) { sPs[wid][r] = ts; sPd[wid][r] = td; sPc[wid][r] = tc; }
  }
  __syncthreads();
  if (threadIdx.x < 16) {
    int r = threadIdx.x;
    float a = 0.f, d = 0.f, cc = 0.f;
    for (int w = 0; w < 4; ++w) { a += sPs[w][r]; d += sPd[w][r]; cc += sPc[w][r]; }
    int gg = b * N_ + i0 + r;
    src2[gg] = a; dst2[gg] = d; colv[gg] = cc;
  }
}

// ---------- layer-2 attention collapsed to online-softmax weighted sum of one channel ----------
__global__ void k_out(const float* __restrict__ src2, const float* __restrict__ dst2,
                      const float* __restrict__ colv, const unsigned* __restrict__ mb,
                      const float* __restrict__ x, float* __restrict__ out) {
  int wid = threadIdx.x >> 5, lane = threadIdx.x & 31;
  int g = blockIdx.x * 8 + wid;
  int b = g >> 10, i = g & (N_ - 1);
  float sv = src2[g];
  const float* dB = dst2 + b * N_;
  const float* vB = colv + b * N_;
  const unsigned* mrow = mb + (size_t)i * (N_ / 32);
  float m = -3.4e38f, s = 0.f, a = 0.f;
  for (int it = 0; it < N_ / 32; ++it) {
    int j = it * 32 + lane;
    float e = sv + dB[j];
    e = e > 0.f ? e : 0.2f * e;
    e = ((mrow[it] >> lane) & 1u) ? e : NEGV;
    float mn = fmaxf(m, e);
    float sc = __expf(m - mn);
    float p = __expf(e - mn);
    s = s * sc + p;
    a = a * sc + p * vB[j];
    m = mn;
  }
#pragma unroll
  for (int off = 16; off >= 1; off >>= 1) {
    float mo = __shfl_xor(m, off, 32);
    float so = __shfl_xor(s, off, 32);
    float ao = __shfl_xor(a, off, 32);
    float mn = fmaxf(m, mo);
    float e1 = __expf(m - mn), e2 = __expf(mo - mn);
    s = s * e1 + so * e2;
    a = a * e1 + ao * e2;
    m = mn;
  }
  if (lane == 0) {
    float r = a / s;
    r = r > 0.f ? r : (__expf(r) - 1.f);
    out[g] = r + x[(size_t)g * F_ + (F_ - 1)];
  }
}

extern "C" void kernel_launch(void* const* d_in, const int* in_sizes, int n_in,
                              void* d_out, int out_size, void* d_ws, size_t ws_size,
                              hipStream_t stream) {
  (void)in_sizes; (void)n_in; (void)out_size; (void)ws_size;
  const float* x     = (const float*)d_in[0];
  const int*   adj   = (const int*)d_in[1];
  const float* W1    = (const float*)d_in[2];
  const float* a1    = (const float*)d_in[3];
  const float* gamma = (const float*)d_in[4];
  const float* beta  = (const float*)d_in[5];
  const float* W2    = (const float*)d_in[6];
  const float* a2    = (const float*)d_in[7];
  float* out = (float*)d_out;

  char* w = (char*)d_ws;
  auto take = [&](size_t bytes) -> char* {
    char* p = w;
    w += (bytes + 255) & ~(size_t)255;
    return p;
  };
  __bf16* xb    = (__bf16*)take((size_t)B_ * N_ * F_ * 2);  // 4 MB
  __bf16* w1t   = (__bf16*)take((size_t)H_ * F_ * 2);
  __bf16* h1bT  = (__bf16*)take((size_t)B_ * H_ * N_ * 2);  // 8 MB
  unsigned* mb  = (unsigned*)take((size_t)N_ * 32 * 4);     // 128 KB
  float* src1   = (float*)take((size_t)B_ * N_ * 4);
  float* dst1   = (float*)take((size_t)B_ * N_ * 4);
  float* m1     = (float*)take((size_t)B_ * N_ * 4);
  float* s1inv  = (float*)take((size_t)B_ * N_ * 4);
  float* src2   = (float*)take((size_t)B_ * N_ * 4);
  float* dst2   = (float*)take((size_t)B_ * N_ * 4);
  float* colv   = (float*)take((size_t)B_ * N_ * 4);
  float* w1s    = (float*)take(F_ * 4);
  float* w1d    = (float*)take(F_ * 4);
  float* w2s    = (float*)take(H_ * 4);
  float* w2d    = (float*)take(H_ * 4);
  float* w2c    = (float*)take(H_ * 4);

  k_prep_vec<<<1, 256, 0, stream>>>(W1, a1, W2, a2, w1s, w1d, w2s, w2d, w2c);
  k_cast_x<<<(B_ * N_ * F_) / 256, 256, 0, stream>>>(x, xb, B_ * N_ * F_);
  k_w1t<<<(H_ * F_) / 256, 256, 0, stream>>>(W1, w1t);
  k_maskpack<<<(N_ * 32) / 256, 256, 0, stream>>>(adj, mb);
  k_srcdst1<<<(B_ * N_) / 8, 256, 0, stream>>>(x, w1s, w1d, src1, dst1);
  k_stats1<<<(B_ * N_) / 8, 256, 0, stream>>>(src1, dst1, mb, m1, s1inv);
  k_gemm1<<<(B_ * (N_ / 16) * (H_ / 16)) / 4, 128, 0, stream>>>(xb, w1t, h1bT);
  k_attn1<<<B_ * (N_ / 16), 128, 0, stream>>>(h1bT, src1, dst1, m1, s1inv, mb,
                                              gamma, beta, w2s, w2d, w2c,
                                              src2, dst2, colv);
  k_out<<<(B_ * N_) / 8, 256, 0, stream>>>(src2, dst2, colv, mb, x, out);
}
